// VectorQuantiser_9440338116774
// MI455X (gfx1250) — matrix operand order repaired and checked by reading, not compiled, not run on hardware
//
#include <hip/hip_runtime.h>

typedef __attribute__((ext_vector_type(16))) _Float16 v16h;
typedef __attribute__((ext_vector_type(8)))  _Float16 v8h;
typedef __attribute__((ext_vector_type(8)))  float    v8f;

#define NUM_EMBED 1024
#define EMBED_DIM 64
#define N_ROWS    131072LL   // 16 * 8192
#define BETA      0.25f

// ---------------------------------------------------------------------------
// Kernel 1: L2-normalize codebook rows, emit f16 codebook [1024][64]
// ---------------------------------------------------------------------------
__global__ void __launch_bounds__(256)
vq_normalize_codebook(const float* __restrict__ w, _Float16* __restrict__ cn)
{
    int row = blockIdx.x * blockDim.x + threadIdx.x;
    if (row >= NUM_EMBED) return;
    const float4* p = (const float4*)(w + (size_t)row * EMBED_DIM);
    float v[EMBED_DIM];
    float ss = 0.f;
#pragma unroll
    for (int i = 0; i < 16; ++i) {
        float4 a = p[i];
        v[4*i+0] = a.x; v[4*i+1] = a.y; v[4*i+2] = a.z; v[4*i+3] = a.w;
        ss += a.x*a.x + a.y*a.y + a.z*a.z + a.w*a.w;
    }
    float inv = 1.0f / fmaxf(sqrtf(ss), 1e-12f);
    _Float16* o = cn + (size_t)row * EMBED_DIM;
#pragma unroll
    for (int i = 0; i < 8; ++i) {
        v8h h;
#pragma unroll
        for (int j = 0; j < 8; ++j) h[j] = (_Float16)(v[8*i+j] * inv);
        *(v8h*)(o + 8*i) = h;
    }
}

// ---------------------------------------------------------------------------
// Kernel 2: main WMMA kernel.
// Block = 256 threads = 8 waves; each wave owns 2 row-tiles (32 z-rows),
// reusing each codebook A-fragment for 2x WMMA pairs (halves DS traffic).
// LDS holds the whole f16 codebook (1024x64 = 128 KB, dynamic).
// D = cn_tile(16xK) x zn^T(Kx16): per-lane accumulator holds 8 codebook
// entries for one z-row -> argmax is per-slot in-lane + one shfl_xor(16).
// ---------------------------------------------------------------------------
__global__ void __launch_bounds__(256)
vq_main_kernel(const float* __restrict__ z, const _Float16* __restrict__ cn,
               int* __restrict__ ws_idx, unsigned int* __restrict__ hist,
               float* __restrict__ out_idx_f)
{
    extern __shared__ _Float16 lds_cn[];   // [1024][64] f16

    const int tid = threadIdx.x;
    // cooperative LDS fill: 1024*64 halfs = 8192 x uint4
    {
        const uint4* src = (const uint4*)cn;
        uint4*       dst = (uint4*)lds_cn;
        for (int i = tid; i < (NUM_EMBED * EMBED_DIM * 2) / 16; i += 256)
            dst[i] = src[i];
    }
    __syncthreads();

    const int lane = tid & 31;
    const int r    = lane & 15;     // z-row within tile / codebook row in tile
    const int half = lane >> 4;     // which K-half this lane carries
    const long long rowBase = (long long)blockIdx.x * 256 + (tid >> 5) * 32;

    // --- load 2 row-tiles of z, normalize, convert to f16 B-fragments ---
    // B fragment (K x N, K=32): lane r holds K[half*16 .. half*16+16) of col r.
    v16h b0[2], b1[2];              // per row-tile: K=0..31 and K=32..63 frags
#pragma unroll
    for (int j = 0; j < 2; ++j) {
        const long long row = rowBase + j * 16 + r;
        float x[32];
        const float4* p0 = (const float4*)(z + row * EMBED_DIM + half * 16);
        const float4* p1 = (const float4*)(z + row * EMBED_DIM + 32 + half * 16);
#pragma unroll
        for (int i = 0; i < 4; ++i) {
            float4 a = p0[i];
            x[4*i+0] = a.x; x[4*i+1] = a.y; x[4*i+2] = a.z; x[4*i+3] = a.w;
        }
#pragma unroll
        for (int i = 0; i < 4; ++i) {
            float4 a = p1[i];
            x[16+4*i+0] = a.x; x[16+4*i+1] = a.y; x[16+4*i+2] = a.z; x[16+4*i+3] = a.w;
        }
        float ss = 0.f;
#pragma unroll
        for (int i = 0; i < 32; ++i) ss += x[i] * x[i];
        ss += __shfl_xor(ss, 16, 32);             // combine the two K-halves
        const float inv = 1.0f / fmaxf(sqrtf(ss), 1e-12f);
#pragma unroll
        for (int i = 0; i < 16; ++i) {
            b0[j][i] = (_Float16)(x[i]      * inv);
            b1[j][i] = (_Float16)(x[16 + i] * inv);
        }
    }

    // --- per-slot running argmax state: value + tile id, per row-tile ---
    float bv[2][8];
    int   bt[2][8];
#pragma unroll
    for (int j = 0; j < 2; ++j)
#pragma unroll
        for (int m = 0; m < 8; ++m) { bv[j][m] = -3.4e38f; bt[j][m] = 0; }

    // --- sweep the 64 codebook tiles; each A-fragment feeds 2 WMMA pairs ---
    for (int t = 0; t < 64; ++t) {
        const _Float16* tb = lds_cn + (t * 16 + r) * EMBED_DIM;
        // A frag (16x32 f16): lane r    -> M=r, K {0..15}
        //                     lane r+16 -> M=r, K {16..31}
        v8h a00 = *(const v8h*)(tb + half * 16);
        v8h a01 = *(const v8h*)(tb + 8 + half * 16);
        v8h a10 = *(const v8h*)(tb + 32 + half * 16);
        v8h a11 = *(const v8h*)(tb + 40 + half * 16);
        v16h a0 = __builtin_shufflevector(a00, a01, 0,1,2,3,4,5,6,7,8,9,10,11,12,13,14,15);
        v16h a1 = __builtin_shufflevector(a10, a11, 0,1,2,3,4,5,6,7,8,9,10,11,12,13,14,15);

        v8f acc0 = {0.f,0.f,0.f,0.f,0.f,0.f,0.f,0.f};
        v8f acc1 = {0.f,0.f,0.f,0.f,0.f,0.f,0.f,0.f};
        acc0 = __builtin_amdgcn_wmma_f32_16x16x32_f16(false, a0, false, b0[0],
                                                      (short)0, acc0, false, false);
        acc0 = __builtin_amdgcn_wmma_f32_16x16x32_f16(false, a1, false, b1[0],
                                                      (short)0, acc0, false, false);
        acc1 = __builtin_amdgcn_wmma_f32_16x16x32_f16(false, a0, false, b0[1],
                                                      (short)0, acc1, false, false);
        acc1 = __builtin_amdgcn_wmma_f32_16x16x32_f16(false, a1, false, b1[1],
                                                      (short)0, acc1, false, false);

        // per-slot tracking: strict > keeps earliest tile on ties
#pragma unroll
        for (int m = 0; m < 8; ++m) {
            float v0 = acc0[m];
            if (v0 > bv[0][m]) { bv[0][m] = v0; bt[0][m] = t; }
            float v1 = acc1[m];
            if (v1 > bv[1][m]) { bv[1][m] = v1; bt[1][m] = t; }
        }
    }

    // --- reduce slots -> (value, global index), then combine lane halves ---
#pragma unroll
    for (int j = 0; j < 2; ++j) {
        float best    = bv[j][0];
        int   bestIdx = bt[j][0] * 16 + half * 8;
#pragma unroll
        for (int m = 1; m < 8; ++m) {
            float v    = bv[j][m];
            int   cand = bt[j][m] * 16 + half * 8 + m;
            if (v > best || (v == best && cand < bestIdx)) { best = v; bestIdx = cand; }
        }
        float ob = __shfl_xor(best, 16, 32);
        int   oi = __shfl_xor(bestIdx, 16, 32);
        if (ob > best || (ob == best && oi < bestIdx)) { best = ob; bestIdx = oi; }

        if (half == 0) {
            long long n = rowBase + j * 16 + r;
            ws_idx[n] = bestIdx;
            atomicAdd(&hist[bestIdx], 1u);
            out_idx_f[n] = (float)bestIdx;
        }
    }
}

// ---------------------------------------------------------------------------
// Kernel 3: gather z_q (= z_q_st numerically) + loss partial sums
// ---------------------------------------------------------------------------
__global__ void __launch_bounds__(256)
vq_gather_loss(const float* __restrict__ z, const float* __restrict__ w,
               const int* __restrict__ ws_idx, float* __restrict__ zq,
               float* __restrict__ loss_acc)
{
    const int gid = blockIdx.x * 256 + threadIdx.x;     // < 8388608
    const int n = gid >> 6, k = gid & 63;
    const int idx = ws_idx[n];
    const float e = w[idx * EMBED_DIM + k];
    const float d = e - z[gid];
    zq[gid] = e;

    float s = d * d;
#pragma unroll
    for (int off = 16; off > 0; off >>= 1) s += __shfl_xor(s, off, 32);

    __shared__ float red[8];
    const int lane = threadIdx.x & 31, wv = threadIdx.x >> 5;
    if (lane == 0) red[wv] = s;
    __syncthreads();
    if (threadIdx.x == 0) {
        float t = 0.f;
#pragma unroll
        for (int i = 0; i < 8; ++i) t += red[i];
        atomicAdd(loss_acc, t);
    }
}

// ---------------------------------------------------------------------------
// Kernel 4: scatter one-hot encodings (region pre-zeroed by memsetAsync)
// ---------------------------------------------------------------------------
__global__ void __launch_bounds__(256)
vq_scatter(const int* __restrict__ ws_idx, float* __restrict__ enc)
{
    const int n = blockIdx.x * 256 + threadIdx.x;
    if (n < (int)N_ROWS) enc[(long long)n * NUM_EMBED + ws_idx[n]] = 1.0f;
}

// ---------------------------------------------------------------------------
// Kernel 5: finalize loss + perplexity
// ---------------------------------------------------------------------------
__global__ void __launch_bounds__(256)
vq_finalize(const unsigned int* __restrict__ hist,
            const float* __restrict__ loss_acc,
            float* __restrict__ out_loss, float* __restrict__ out_perp)
{
    float s = 0.f;
    for (int i = threadIdx.x; i < NUM_EMBED; i += 256) {
        float p = (float)hist[i] * (1.0f / (float)N_ROWS);
        s += p * logf(p + 1e-10f);
    }
#pragma unroll
    for (int off = 16; off > 0; off >>= 1) s += __shfl_xor(s, off, 32);
    __shared__ float red[8];
    const int lane = threadIdx.x & 31, wv = threadIdx.x >> 5;
    if (lane == 0) red[wv] = s;
    __syncthreads();
    if (threadIdx.x == 0) {
        float t = 0.f;
#pragma unroll
        for (int i = 0; i < 8; ++i) t += red[i];
        *out_perp = expf(-t);
        *out_loss = (1.0f + BETA) * (*loss_acc) / ((float)N_ROWS * (float)EMBED_DIM);
    }
}

// ---------------------------------------------------------------------------
extern "C" void kernel_launch(void* const* d_in, const int* in_sizes, int n_in,
                              void* d_out, int out_size, void* d_ws, size_t ws_size,
                              hipStream_t stream)
{
    (void)in_sizes; (void)n_in; (void)out_size; (void)ws_size;

    const float* z = (const float*)d_in[0];      // [16,8192,64] f32
    const float* w = (const float*)d_in[1];      // [1024,64]   f32
    float* out = (float*)d_out;

    // workspace layout
    char* ws = (char*)d_ws;
    _Float16*     cn       = (_Float16*)ws;                       // 131072 B
    int*          idxbuf   = (int*)(ws + 131072);                 // 524288 B
    unsigned int* hist     = (unsigned int*)(ws + 655360);        //   4096 B
    float*        loss_acc = (float*)(ws + 659456);               //      4 B

    // output layout (flat, return order)
    float* zq    = out;                                           // 8388608
    float* oLoss = out + 8388608LL;                               // 1
    float* oPerp = out + 8388609LL;                               // 1
    float* enc   = out + 8388610LL;                               // 134217728
    float* oIdx  = out + 8388610LL + 134217728LL;                 // 131072

    hipMemsetAsync(hist, 0, 4096 + 4, stream);                    // hist + loss_acc
    hipMemsetAsync(enc, 0, 134217728ULL * sizeof(float), stream); // one-hot zeros

    vq_normalize_codebook<<<NUM_EMBED / 256, 256, 0, stream>>>(w, cn);

    // 512 blocks x 256 threads; 128 KB dynamic LDS (f16 codebook)
    vq_main_kernel<<<(int)(N_ROWS / 256), 256,
                     NUM_EMBED * EMBED_DIM * sizeof(_Float16), stream>>>(
        z, cn, idxbuf, hist, oIdx);

    vq_gather_loss<<<(int)(N_ROWS * EMBED_DIM / 256), 256, 0, stream>>>(
        z, w, idxbuf, zq, loss_acc);

    vq_scatter<<<(int)(N_ROWS / 256), 256, 0, stream>>>(idxbuf, enc);

    vq_finalize<<<1, 256, 0, stream>>>(hist, loss_acc, oLoss, oPerp);
}
